// TSEncoder_17523466567872
// MI455X (gfx1250) — matrix operand order repaired and checked by reading, not compiled, not run on hardware
//
#include <hip/hip_runtime.h>

typedef __attribute__((ext_vector_type(16))) _Float16 v16h;
typedef __attribute__((ext_vector_type(8)))  _Float16 v8h;
typedef __attribute__((ext_vector_type(8)))  float    v8f;
typedef __attribute__((ext_vector_type(4)))  float    v4f;
typedef __attribute__((ext_vector_type(2)))  float    v2f;

#define BB   4096
#define TT   250
#define DD   4
#define HH   64
#define NG   256          // 4*H gate outputs
#define TOUT 50           // T/5 output steps

// workspace layout (in floats)
#define WS_SUM   0                 // [4]   per-channel sum
#define WS_SQ    4                 // [4]   per-channel sum of squares
#define WS_BIAS  8                 // [256] fused bias b' = b_ih+b_hh+W_ih@shift
#define WS_WP    (8 + 256)         // [256*4] fused input weights W'
#define WS_WHH   (8 + 256 + 1024)  // f16 region: 256*64 halves of W_hh

// ---------------------------------------------------- fast activations -----
// Branch-free, on the TRANS pipe. sigmoid(x) = rcp(1 + 2^(-x*log2e)).
__device__ __forceinline__ float sigm(float x) {
    return __builtin_amdgcn_rcpf(1.0f + __builtin_amdgcn_exp2f(-1.44269504f * x));
}
__device__ __forceinline__ float ftanh(float x) {
#if __has_builtin(__builtin_amdgcn_tanhf)
    return __builtin_amdgcn_tanhf(x);            // CDNA5 v_tanh_f32
#elif __has_builtin(__builtin_amdgcn_tanh_f32)
    return __builtin_amdgcn_tanh_f32(x);
#else
    // tanh(x) = 1 - 2/(2^(2x*log2e) + 1)
    return 1.0f - 2.0f * __builtin_amdgcn_rcpf(1.0f + __builtin_amdgcn_exp2f(2.88539008f * x));
#endif
}

// ---------------------------------------------------------------- stats ----
__global__ void k_zero(float* ws) {
    if (threadIdx.x < 8) ws[threadIdx.x] = 0.0f;
}

__global__ void k_stats(const float* __restrict__ x, float* __restrict__ ws) {
    float s0 = 0, s1 = 0, s2 = 0, s3 = 0;
    float q0 = 0, q1 = 0, q2 = 0, q3 = 0;
    int stride = gridDim.x * blockDim.x;
    for (int i = blockIdx.x * blockDim.x + threadIdx.x; i < BB * TT; i += stride) {
        v4f v = *(const v4f*)(x + (size_t)i * 4);
        s0 += v.x; q0 += v.x * v.x;
        s1 += v.y; q1 += v.y * v.y;
        s2 += v.z; q2 += v.z * v.z;
        s3 += v.w; q3 += v.w * v.w;
    }
#pragma unroll
    for (int off = 16; off > 0; off >>= 1) {
        s0 += __shfl_down(s0, off, 32); q0 += __shfl_down(q0, off, 32);
        s1 += __shfl_down(s1, off, 32); q1 += __shfl_down(q1, off, 32);
        s2 += __shfl_down(s2, off, 32); q2 += __shfl_down(q2, off, 32);
        s3 += __shfl_down(s3, off, 32); q3 += __shfl_down(q3, off, 32);
    }
    if ((threadIdx.x & 31) == 0) {
        atomicAdd(&ws[WS_SUM + 0], s0); atomicAdd(&ws[WS_SQ + 0], q0);
        atomicAdd(&ws[WS_SUM + 1], s1); atomicAdd(&ws[WS_SQ + 1], q1);
        atomicAdd(&ws[WS_SUM + 2], s2); atomicAdd(&ws[WS_SQ + 2], q2);
        atomicAdd(&ws[WS_SUM + 3], s3); atomicAdd(&ws[WS_SQ + 3], q3);
    }
}

// ------------------------------------------------------------------ prep ---
// Fold BatchNorm into the input projection; convert W_hh to f16.
__global__ void k_prep(const float* __restrict__ gamma, const float* __restrict__ beta,
                       const float* __restrict__ W_ih,  const float* __restrict__ W_hh,
                       const float* __restrict__ b_ih,  const float* __restrict__ b_hh,
                       float* __restrict__ ws) {
    int n = threadIdx.x;            // 0..255 gate row
    const float Ninv = 1.0f / ((float)BB * (float)TT);
    float scale[DD], shift[DD];
#pragma unroll
    for (int d = 0; d < DD; ++d) {
        float mean = ws[WS_SUM + d] * Ninv;
        float var  = ws[WS_SQ + d] * Ninv - mean * mean;   // biased, like jnp.var
        float a    = __frsqrt_rn(var + 1e-5f) * gamma[d];
        scale[d] = a;
        shift[d] = beta[d] - mean * a;
    }
    float bb = b_ih[n] + b_hh[n];
#pragma unroll
    for (int d = 0; d < DD; ++d) {
        float w = W_ih[n * DD + d];
        ws[WS_WP + n * DD + d] = w * scale[d];
        bb += w * shift[d];
    }
    ws[WS_BIAS + n] = bb;
    _Float16* whhT = (_Float16*)(ws + WS_WHH);
#pragma unroll
    for (int k = 0; k < HH; ++k)
        whhT[n * HH + k] = (_Float16)W_hh[n * HH + k];     // column n of B, K contiguous
}

// ------------------------------------------------------------------ LSTM ---
// One 16-row batch tile per workgroup, split across 4 waves: wave j owns
// hidden slice 16j..16j+15 (gate tiles tn = q*4+j, q=i,f,g,o). Each wave:
// 12 WMMAs + 8 elems/lane of activations per step — slices are independent,
// so the only cross-wave coupling is h itself, exchanged through a
// double-buffered LDS tile with ONE barrier per step. W_hh fragments are
// register-resident (only 64 VGPRs now; whole wave fits under 256 VGPRs).
#define LDSPAD 72   // 64 halves + 16B pad per row (keeps 16B alignment, rotates banks)

__global__ void __launch_bounds__(128)
k_lstm(const float* __restrict__ x, const float* __restrict__ ws,
       float* __restrict__ out) {
    __shared__ _Float16 hbuf[2][16 * LDSPAD];     // double-buffered h (4.6 KB)

    const int lane = threadIdx.x & 31;
    const int j    = threadIdx.x >> 5;            // wave index == hidden slice
    const int lo   = lane & 15;                   // N-index / row-index within tile
    const int hi   = lane >> 4;                   // half-wave select

    const int tileBase = blockIdx.x * 16;
    // x A-fragment (f32 16x4): lane lo = row M, VGPR pair holds K = 2*hi, 2*hi+1
    const float* xrow = x + (size_t)(tileBase + lo) * TT * DD + 2 * hi;

    // ---- loop-invariant operands for this wave's 4 gate tiles ----
    const _Float16* whhT = (const _Float16*)(ws + WS_WHH);
    v16h  whB[4][2];
    v2f   wpb[4];
    float biasv[4];
#pragma unroll
    for (int q = 0; q < 4; ++q) {
        const int tn = q * 4 + j;
        const int n  = tn * 16 + lo;
        biasv[q] = ws[WS_BIAS + n];
        wpb[q]   = *(const v2f*)(ws + WS_WP + n * DD + 2 * hi);
#pragma unroll
        for (int kc = 0; kc < 2; ++kc) {
            const _Float16* bp = whhT + n * HH + kc * 32 + hi * 16;
            *((v8h*)&whB[q][kc])     = *(const v8h*)(bp);
            *((v8h*)&whB[q][kc] + 1) = *(const v8h*)(bp + 8);
        }
    }

    // state: this wave's hidden-slice cell state + full-h A fragments
    v8f cj;
    v16h hA0, hA1;
#pragma unroll
    for (int r = 0; r < 8; ++r) cj[r] = 0.0f;
#pragma unroll
    for (int i = 0; i < 16; ++i) { hA0[i] = (_Float16)0.0f; hA1[i] = (_Float16)0.0f; }

    // software-prefetched x fragment (t=0 reads flipped t_src = TT-1)
    v2f xa = *(const v2f*)(xrow + (TT - 1) * DD);

    int tmod = 0, j5 = 0;   // incremental t%5 and t/5 trackers

#pragma clang loop unroll(disable)
    for (int t = 0; t < TT; ++t) {
        // prefetch next step's x fragment (overlaps this step's compute)
        const int tsrc_n = (t + 1 < TT) ? (TT - 2 - t) : 0;
        v2f xa_next = *(const v2f*)(xrow + tsrc_n * DD);

        const bool emit = (tmod == 4);
        _Float16* hb = hbuf[t & 1];

        // gates for this hidden slice: i, f, g, o
        v8f g[4];
#pragma unroll
        for (int q = 0; q < 4; ++q) {
            v8f acc;
#pragma unroll
            for (int r = 0; r < 8; ++r) acc[r] = biasv[q];
            acc = __builtin_amdgcn_wmma_f32_16x16x4_f32(
                    false, xa, false, wpb[q], (short)0, acc, false, false);
            acc = __builtin_amdgcn_wmma_f32_16x16x32_f16(
                    false, hA0, false, whB[q][0], (short)0, acc, false, false);
            acc = __builtin_amdgcn_wmma_f32_16x16x32_f16(
                    false, hA1, false, whB[q][1], (short)0, acc, false, false);
            g[q] = acc;
        }

        // elementwise cell update (branch-free, 8 elems/lane)
#pragma unroll
        for (int r = 0; r < 8; ++r) {
            float iv = sigm(g[0][r]);
            float fv = sigm(g[1][r]);
            float gv = ftanh(g[2][r]);
            float ov = sigm(g[3][r]);
            float c  = fv * cj[r] + iv * gv;
            cj[r] = c;
            float hv = ov * ftanh(c);
            const int M = r + 8 * hi;               // C layout: row = r + 8*hi
            hb[M * LDSPAD + j * 16 + lo] = (_Float16)hv;
            if (emit)
                out[(size_t)(tileBase + M) * (TOUT * HH) + j5 * HH + j * 16 + lo] = hv;
        }

        __syncthreads();   // publish h slice; buffer parity handles WAR

        // rebuild full-h A-fragments (f16 16x32 layout) from LDS
        {
            const _Float16* hp = hb + lo * LDSPAD;  // row M = lo
            *((v8h*)&hA0)     = *(const v8h*)(hp + hi * 16);       // K = 16*hi .. +7
            *((v8h*)&hA0 + 1) = *(const v8h*)(hp + 8 + hi * 16);   // K = 8+16*hi ..
            *((v8h*)&hA1)     = *(const v8h*)(hp + 32 + hi * 16);  // K = 32+16*hi ..
            *((v8h*)&hA1 + 1) = *(const v8h*)(hp + 40 + hi * 16);  // K = 40+16*hi ..
        }
        xa = xa_next;
        j5  += emit ? 1 : 0;
        tmod = emit ? 0 : tmod + 1;
    }
}

// ---------------------------------------------------------------- launch ---
extern "C" void kernel_launch(void* const* d_in, const int* in_sizes, int n_in,
                              void* d_out, int out_size, void* d_ws, size_t ws_size,
                              hipStream_t stream) {
    const float* x     = (const float*)d_in[0];
    const float* gamma = (const float*)d_in[1];
    const float* beta  = (const float*)d_in[2];
    const float* W_ih  = (const float*)d_in[3];
    const float* W_hh  = (const float*)d_in[4];
    const float* b_ih  = (const float*)d_in[5];
    const float* b_hh  = (const float*)d_in[6];
    float* out = (float*)d_out;
    float* ws  = (float*)d_ws;

    k_zero <<<1, 32, 0, stream>>>(ws);
    k_stats<<<512, 256, 0, stream>>>(x, ws);
    k_prep <<<1, 256, 0, stream>>>(gamma, beta, W_ih, W_hh, b_ih, b_hh, ws);
    k_lstm <<<BB / 16, 128, 0, stream>>>(x, ws, out);
}
